// NeuralCDE_60430189855046
// MI455X (gfx1250) — matrix operand order repaired and checked by reading, not compiled, not run on hardware
//
#include <hip/hip_runtime.h>

typedef __attribute__((ext_vector_type(16))) _Float16 v16h;
typedef __attribute__((ext_vector_type(2)))  _Float16 v2h;
typedef __attribute__((ext_vector_type(2)))  __fp16   v2hf;
typedef __attribute__((ext_vector_type(8)))  float    v8f;

#define B_ 4096
#define T_ 2048
#define I_ 4
#define H_ 32
#define W_ 128

#define WAVES   2
#define THREADS (WAVES * 32)

// Hardware tanh (CDNA5 transcendental). Builtin if available, else raw ISA.
__device__ __forceinline__ float fast_tanh(float x) {
#if __has_builtin(__builtin_amdgcn_tanhf)
    return __builtin_amdgcn_tanhf(x);
#else
    float y;
    asm("v_tanh_f32 %0, %1\n\tv_nop" : "=v"(y) : "v"(x));
    return y;
#endif
}

// Single-instruction ReLU: med3(x, 0, big) == clamp(x, 0, big).
__device__ __forceinline__ float relu_f(float x) {
#if __has_builtin(__builtin_amdgcn_fmed3f)
    return __builtin_amdgcn_fmed3f(x, 0.0f, 3.0e38f);
#else
    return x > 0.f ? x : 0.f;
#endif
}

// Pack two f32 into f16x2 with one v_cvt_pk_rtz_f16_f32.
__device__ __forceinline__ v2h pk_f16(float a, float b) {
#if __has_builtin(__builtin_amdgcn_cvt_pkrtz)
    v2hf t = __builtin_amdgcn_cvt_pkrtz(a, b);
    return __builtin_bit_cast(v2h, t);
#else
    v2h r; r[0] = (_Float16)a; r[1] = (_Float16)b; return r;
#endif
}

// Transposed persistent Neural-CDE kernel.
// One wave owns 16 batch rows (batch lives on LANES as the WMMA N axis).
// GEMM1: h^T[128,16] = W1^T[128,32] @ z^T[32,16]   -> 8  x v_wmma_f32_16x16x32_f16
// GEMM2: F^T[128,16] = W2^T[128,128] @ h^T[128,16] -> 32 x v_wmma_f32_16x16x32_f16
// All weights + bias C-fragments live in the wave's (1024-deep) VGPR file.
__global__ __launch_bounds__(THREADS, 1)
void ncde_kernel(const float* __restrict__ X,  const float* __restrict__ Wi,
                 const float* __restrict__ bi, const float* __restrict__ W1,
                 const float* __restrict__ b1, const float* __restrict__ W2,
                 const float* __restrict__ b2, const float* __restrict__ Wo,
                 const float* __restrict__ bo, float* __restrict__ out)
{
    // Staging for the one-time weight swizzle (A-fragment layout, contiguous per lane).
    __shared__ __align__(32) _Float16 w1p[8 * 32 * 16];       // W1^T tiles   8 KB
    __shared__ __align__(32) _Float16 w2p[4 * 8 * 32 * 16];   // W2^T tiles  32 KB
    __shared__ __align__(32) float    b1s[W_];
    __shared__ __align__(32) float    b2s[W_];
    // Wave-private staging (no cross-wave sharing -> no barriers in the loop).
    __shared__ __align__(32) _Float16 zbufB[WAVES][32 * 16];      // z^T  B-frag
    __shared__ __align__(32) _Float16 hbufB[WAVES][4 * 32 * 16];  // h^T  B-frag (4 K-chunks)
    __shared__ __align__(16) float    dxs[WAVES][16 * 4];

    const int tid  = threadIdx.x;
    const int lane = tid & 31;
    const int wv   = tid >> 5;
    const int ln   = lane & 15;     // batch index within the wave tile
    const int half = lane >> 4;

    // ---- pack W1^T / W2^T into A-fragment order (once per block) ----
    // A-frag element j of lane s: m = s&15, k' = j + (s>>4)*16
    for (int idx = tid; idx < 8 * 32 * 16; idx += THREADS) {
        int j = idx & 15, s = (idx >> 4) & 31, mt = idx >> 9;
        int v = j >> 1, hh = j & 1;
        int kp = (v * 2 + hh) + ((s >> 4) * 16);                          // K (H dim, 0..31)
        int fr = mt * 16 + (s & 15);                                      // feature row (W dim)
        w1p[idx] = (_Float16)W1[kp * W_ + fr];                            // W1^T[fr][kp]
    }
    for (int idx = tid; idx < 4 * 8 * 32 * 16; idx += THREADS) {
        int j = idx & 15, s = (idx >> 4) & 31, mt = (idx >> 9) & 7, kk = idx >> 12;
        int v = j >> 1, hh = j & 1;
        int kp = (v * 2 + hh) + ((s >> 4) * 16);                          // K within chunk
        int fr = mt * 16 + (s & 15);                                      // out-feature row
        w2p[idx] = (_Float16)W2[(kk * 32 + kp) * W_ + fr];                // W2^T[fr][k]
    }
    for (int idx = tid; idx < W_; idx += THREADS) { b1s[idx] = b1[idx]; b2s[idx] = b2[idx]; }
    __syncthreads();

    // ---- registerize everything loop-invariant ----
    v16h w1f[8];                      //  64 VGPRs: W1^T A-fragments
#pragma unroll
    for (int mt = 0; mt < 8; ++mt) w1f[mt] = *(const v16h*)&w1p[(mt * 32 + lane) * 16];
    v16h w2f[32];                     // 256 VGPRs: W2^T A-fragments [kk*8+mt]
#pragma unroll
    for (int q = 0; q < 32; ++q) w2f[q] = *(const v16h*)&w2p[(q * 32 + lane) * 16];
    v8f b1f[8], b2f[8];               // 128 VGPRs: bias C-fragments
#pragma unroll
    for (int mt = 0; mt < 8; ++mt) {
        b1f[mt] = *(const v8f*)&b1s[mt * 16 + half * 8];
        b2f[mt] = *(const v8f*)&b2s[mt * 16 + half * 8];
    }

    const int rowBase = (blockIdx.x * WAVES + wv) * 16;
    const int bidx    = rowBase + ln;     // this lane's batch row

    // ---- state layout: z[idx] holds z[bidx][h], h = (idx>>1)*4 + half*2 + (idx&1) ----
    float z[16];
    {
        float xv[I_];
#pragma unroll
        for (int i = 0; i < I_; ++i) xv[i] = X[((size_t)bidx * T_) * I_ + i];
#pragma unroll
        for (int idx = 0; idx < 16; ++idx) {
            int h = (idx >> 1) * 4 + half * 2 + (idx & 1);
            float a = bi[h];
#pragma unroll
            for (int i = 0; i < I_; ++i) a += xv[i] * Wi[i * H_ + h];
            z[idx] = a;
        }
    }

    // ---- streaming dX (each lane carries 2 of the 16x4 tile), pipelined ----
    const int r  = lane >> 1;
    const int i0 = (lane & 1) * 2;
    const float* xptr = X + ((size_t)(rowBase + r) * T_) * I_ + i0;
    float2 xc = *(const float2*)(xptr);
    float2 xn = *(const float2*)(xptr + I_);

    float dxl[4];   // dx[bidx][0..3]

    // ---- vector field: dz = tanh(relu(z W1 + b1) W2 + b2) . dx  (all transposed) ----
    auto vf = [&](const float* zin, float* dz) {
        _Float16* zb = zbufB[wv];
        // scatter state into z^T B-fragment layout; pairs are LDS-adjacent -> b32 stores
#pragma unroll
        for (int m = 0; m < 8; ++m) {
            int off = (ln + (m >= 4 ? 16 : 0)) * 16 + ((m * 4 + half * 2) & 15);
            *(v2h*)&zb[off] = pk_f16(zin[2 * m], zin[2 * m + 1]);
        }
        v16h zB = *(const v16h*)&zb[lane * 16];

        _Float16* hb = hbufB[wv];
        // GEMM1: C preloaded with b1 fragment; ReLU; packed scatter of h^T B-frags
#pragma unroll
        for (int mt = 0; mt < 8; ++mt) {
            v8f c = __builtin_amdgcn_wmma_f32_16x16x32_f16(false, w1f[mt], false, zB,
                                                           (short)0, b1f[mt], false, false);
            int ob = ((mt >> 1) * 32 + ln + (mt & 1) * 16) * 16 + half * 8;
#pragma unroll
            for (int p = 0; p < 4; ++p)
                *(v2h*)&hb[ob + 2 * p] = pk_f16(relu_f(c[2 * p]), relu_f(c[2 * p + 1]));
        }
        v16h hB[4];
#pragma unroll
        for (int kk = 0; kk < 4; ++kk) hB[kk] = *(const v16h*)&hb[(kk * 32 + lane) * 16];

        // GEMM2 (K=128, W2 resident in VGPRs); tanh; per-lane contraction over I=4
#pragma unroll
        for (int mt = 0; mt < 8; ++mt) {
            v8f c = __builtin_amdgcn_wmma_f32_16x16x32_f16(false, w2f[mt], false, hB[0],
                                                           (short)0, b2f[mt], false, false);
#pragma unroll
            for (int kk = 1; kk < 4; ++kk)
                c = __builtin_amdgcn_wmma_f32_16x16x32_f16(false, w2f[kk * 8 + mt], false, hB[kk],
                                                           (short)0, c, false, false);
            float t0 = fast_tanh(c[0]) * dxl[0];
            t0 = fmaf(fast_tanh(c[1]), dxl[1], t0);
            t0 = fmaf(fast_tanh(c[2]), dxl[2], t0);
            t0 = fmaf(fast_tanh(c[3]), dxl[3], t0);
            float t1 = fast_tanh(c[4]) * dxl[0];
            t1 = fmaf(fast_tanh(c[5]), dxl[1], t1);
            t1 = fmaf(fast_tanh(c[6]), dxl[2], t1);
            t1 = fmaf(fast_tanh(c[7]), dxl[3], t1);
            dz[mt * 2]     = t0;    // h = mt*4 + half*2 + 0
            dz[mt * 2 + 1] = t1;    // h = mt*4 + half*2 + 1
        }
    };

    // ---- main recurrence: one RK4 step per observation interval ----
    for (int t = 1; t < T_; ++t) {
        float2 dx;
        dx.x = xn.x - xc.x;
        dx.y = xn.y - xc.y;
        xc = xn;
        int tn = (t + 1 < T_) ? (t + 1) : t;              // prefetch next interval
        xn = *(const float2*)(xptr + (size_t)tn * I_);

        dxs[wv][r * 4 + i0]     = dx.x;
        dxs[wv][r * 4 + i0 + 1] = dx.y;
        {
            float4 d4 = *(const float4*)&dxs[wv][ln * 4];
            dxl[0] = d4.x; dxl[1] = d4.y; dxl[2] = d4.z; dxl[3] = d4.w;
        }

        float kd[16], tin[16], acc[16];
        vf(z, kd);                                         // k1
#pragma unroll
        for (int i = 0; i < 16; ++i) { acc[i] = kd[i]; tin[i] = z[i] + 0.5f * kd[i]; }
        vf(tin, kd);                                       // k2
#pragma unroll
        for (int i = 0; i < 16; ++i) { acc[i] += 2.f * kd[i]; tin[i] = z[i] + 0.5f * kd[i]; }
        vf(tin, kd);                                       // k3
#pragma unroll
        for (int i = 0; i < 16; ++i) { acc[i] += 2.f * kd[i]; tin[i] = z[i] + kd[i]; }
        vf(tin, kd);                                       // k4
#pragma unroll
        for (int i = 0; i < 16; ++i) z[i] += (acc[i] + kd[i]) * (1.f / 6.f);
    }

    // ---- readout: out[b] = z[b,:] @ W_out + b_out (reduce across lane halves) ----
    float o = 0.f;
#pragma unroll
    for (int idx = 0; idx < 16; ++idx) {
        int h = (idx >> 1) * 4 + half * 2 + (idx & 1);
        o += z[idx] * Wo[h];
    }
    o += __shfl_xor(o, 16, 32);
    if (lane < 16) out[bidx] = o + bo[0];
}

extern "C" void kernel_launch(void* const* d_in, const int* in_sizes, int n_in,
                              void* d_out, int out_size, void* d_ws, size_t ws_size,
                              hipStream_t stream) {
    (void)in_sizes; (void)n_in; (void)d_ws; (void)ws_size; (void)out_size;
    const float* X  = (const float*)d_in[0];
    const float* Wi = (const float*)d_in[1];
    const float* bi = (const float*)d_in[2];
    const float* W1 = (const float*)d_in[3];
    const float* b1 = (const float*)d_in[4];
    const float* W2 = (const float*)d_in[5];
    const float* b2 = (const float*)d_in[6];
    const float* Wo = (const float*)d_in[7];
    const float* bo = (const float*)d_in[8];
    float* out = (float*)d_out;

    const int grid = B_ / (16 * WAVES);   // 128 blocks x 2 waves = 256 waves
    ncde_kernel<<<grid, THREADS, 0, stream>>>(X, Wi, bi, W1, b1, W2, b2, Wo, bo, out);
}
